// FNOBlock_20263655702999
// MI455X (gfx1250) — matrix operand order repaired and checked by reading, not compiled, not run on hardware
//
#include <hip/hip_runtime.h>
#include <hip/hip_bf16.h>

// ---------------------------------------------------------------------------
// MI455X (gfx1250) FNO block. wave32, WMMA f32_16x16x32_f16 for all GEMMs
// (1x1 convs, DFT-as-GEMM, SSD projections, chunked SSD scan, MLP).
// B operands are staged TRANSPOSED in LDS so each lane's 16-element fragment
// is one contiguous 32B load (2x ds_load_b128); A fragments are 2x 16B loads.
// ---------------------------------------------------------------------------

typedef __attribute__((ext_vector_type(16))) _Float16 v16h;
typedef __attribute__((ext_vector_type(8)))  _Float16 v8h;
typedef __attribute__((ext_vector_type(8)))  float    v8f;

#define NB   4      // batch
#define NC   64     // channels
#define NH   128
#define NW   128
#define NHF  128
#define NWF  65     // rfft width
#define WFP  80     // padded freq width (multiple of 16)
#define LSEQ 8320   // NHF*NWF
#define ND   128    // SSD model dim
#define NN   64     // SSD state dim
#define NR   33280  // NB*LSEQ rows
#define INV_SQRT_N 0.0078125f   // 1/sqrt(H*W) = 1/128
#define SQRT_N     128.0f
#define ANG128 0.049087385212f  // 2*pi/128
#define ISQ128 0.088388347648f  // 1/sqrt(128)

// ----- WMMA fragment builders (CDNA5 ISA 7.12.2, wave32) ---------
// A (16xK row-major f16 LDS): lane m=lane&15; k = i + 16*(lane>=16)
__device__ __forceinline__ v16h frag_a(const _Float16* A, int lda, int ks) {
  const int lane = threadIdx.x & 31;
  const int m = lane & 15;
  const int khi = (lane >> 4) << 4;
  const _Float16* p = A + m * lda + ks * 32 + khi;
  v8h lo = *(const v8h*)p;          // k: base .. base+7
  v8h hi = *(const v8h*)(p + 8);    // k: base+8 .. base+15
  return __builtin_shufflevector(lo, hi, 0,1,2,3,4,5,6,7,8,9,10,11,12,13,14,15);
}
// A generated analytically (DFT twiddles): f(m_global, k_global)
template <typename F>
__device__ __forceinline__ v16h frag_a_tab(F f, int mBase, int ks) {
  const int lane = threadIdx.x & 31;
  const int m = mBase + (lane & 15);
  const int khi = (lane >> 4) << 4;
  v16h a;
#pragma unroll
  for (int i = 0; i < 16; ++i) {
    int k = ks * 32 + i + khi;
    a[i] = (_Float16)f(m, k);
  }
  return a;
}
// B staged transposed [n][k] f16 LDS: lane n=lane&15; k = i + 16*(lane>=16)
// -> 16 contiguous halves = one 32B load.
__device__ __forceinline__ v16h frag_bT(const _Float16* BT, int ldk, int ks) {
  const int lane = threadIdx.x & 31;
  const int n = lane & 15;
  const int khi = (lane >> 4) << 4;
  return *(const v16h*)(BT + n * ldk + ks * 32 + khi);
}
// B transposed f32 LDS (running SSD state), convert per load
__device__ __forceinline__ v16h frag_bT_f32(const float* BT, int ldk, int ks) {
  const int lane = threadIdx.x & 31;
  const int n = lane & 15;
  const int khi = (lane >> 4) << 4;
  const float* p = BT + n * ldk + ks * 32 + khi;
  v16h b;
#pragma unroll
  for (int i = 0; i < 16; ++i) b[i] = (_Float16)p[i];
  return b;
}
// B generated analytically: f(k_global, n_global)
template <typename F>
__device__ __forceinline__ v16h frag_b_tab(F f, int nBase, int ks) {
  const int lane = threadIdx.x & 31;
  const int n = nBase + (lane & 15);
  const int khi = (lane >> 4) << 4;
  v16h b;
#pragma unroll
  for (int i = 0; i < 16; ++i) b[i] = (_Float16)f(ks * 32 + khi + i, n);
  return b;
}

#define WMMA(a, b, acc) \
  acc = __builtin_amdgcn_wmma_f32_16x16x32_f16(false, a, false, b, (short)0, acc, false, false)

// acc += A(row-major f16) @ B(transposed f16)
__device__ __forceinline__ void wmma_rr(const _Float16* A, int lda,
                                        const _Float16* BT, int ldk,
                                        int ksteps, v8f& acc) {
  for (int ks = 0; ks < ksteps; ++ks) {
    v16h a = frag_a(A, lda, ks);
    v16h b = frag_bT(BT, ldk, ks);
    WMMA(a, b, acc);
  }
}
// acc += A(row-major f16) @ B(transposed f32)
__device__ __forceinline__ void wmma_rf(const _Float16* A, int lda,
                                        const float* BT, int ldk,
                                        int ksteps, v8f& acc) {
  for (int ks = 0; ks < ksteps; ++ks) {
    v16h a = frag_a(A, lda, ks);
    v16h b = frag_bT_f32(BT, ldk, ks);
    WMMA(a, b, acc);
  }
}
// acc += Tab @ B(transposed f16)
template <typename F>
__device__ __forceinline__ void wmma_tr(F f, int mBase,
                                        const _Float16* BT, int ldk,
                                        int ksteps, v8f& acc) {
  for (int ks = 0; ks < ksteps; ++ks) {
    v16h a = frag_a_tab(f, mBase, ks);
    v16h b = frag_bT(BT, ldk, ks);
    WMMA(a, b, acc);
  }
}
// acc += A(row-major f16) @ Tab
template <typename F>
__device__ __forceinline__ void wmma_rt(const _Float16* A, int lda,
                                        F f, int nBase,
                                        int ksteps, v8f& acc) {
  for (int ks = 0; ks < ksteps; ++ks) {
    v16h a = frag_a(A, lda, ks);
    v16h b = frag_b_tab(f, nBase, ks);
    WMMA(a, b, acc);
  }
}

// DFT twiddle functors (symmetric 128-point matrices, ortho-normalized)
struct FCos { __device__ float operator()(int a, int k) const {
  return __cosf(ANG128 * (float)((a * k) & 127)) * ISQ128; } };
struct FSin { __device__ float operator()(int a, int k) const {
  return __sinf(ANG128 * (float)((a * k) & 127)) * ISQ128; } };
// inverse-rfft column matrices, Hermitian weights, zero past wf=65
struct FIwc { __device__ float operator()(int k, int n) const {
  if (k >= NWF) return 0.f;
  float w = (k == 0 || k == 64) ? 1.f : 2.f;
  return w * __cosf(ANG128 * (float)((k * n) & 127)) * ISQ128; } };
struct FIws { __device__ float operator()(int k, int n) const {
  if (k >= NWF) return 0.f;
  float w = (k == 0 || k == 64) ? 1.f : 2.f;
  return w * __sinf(ANG128 * (float)((k * n) & 127)) * ISQ128; } };

__device__ __forceinline__ float gelu_exact(float v) {
  return 0.5f * v * (1.f + erff(v * 0.70710678118f));
}

// ---------------------------------------------------------------------------
// weights -> f16 scratch copies (B-side matrices stored TRANSPOSED [n][k])
__global__ void k_prep(const float* pw_w, const float* W_B, const float* W_C,
                       const float* W_out, const float* w1, const float* w2,
                       _Float16* pw16, _Float16* wB16t, _Float16* wC16t,
                       _Float16* wo16t, _Float16* w1_16, _Float16* w2_16) {
  int i = blockIdx.x * 256 + threadIdx.x;
  if (i < 64 * 64)   pw16[i]  = (_Float16)pw_w[i];      // A-side, row-major
  if (i < 128 * 64)  {
    int k = i >> 6, n = i & 63;
    wB16t[n * 128 + k] = (_Float16)W_B[i];              // B-side, [n][k]
    wC16t[n * 128 + k] = (_Float16)W_C[i];
    w1_16[i] = (_Float16)w1[i];                         // A-side
    w2_16[i] = (_Float16)w2[i];                         // A-side
  }
  if (i < 128 * 128) {
    int k = i >> 7, n = i & 127;
    wo16t[n * 128 + k] = (_Float16)W_out[i];            // B-side, [n][k]
  }
}

// x1 = pw_w @ x + pw_b  (per-b GEMM 64x64 x 64x16384)
__global__ void k_pw(const float* x, const _Float16* pw16, const float* pw_b,
                     float* X1) {
  const int b = blockIdx.y, p0 = blockIdx.x * 256;
  const int tid = threadIdx.x, wave = tid >> 5, lane = tid & 31;
  alignas(32) __shared__ _Float16 A16[64 * 64];
  alignas(32) __shared__ _Float16 Bt16[256 * 64];       // [p][c]
  for (int i = tid; i < 64 * 64; i += 256) A16[i] = pw16[i];
  for (int i = tid; i < 64 * 256; i += 256) {
    int c = i >> 8, p = i & 255;   // consecutive tid -> consecutive p (coalesced)
    Bt16[p * 64 + c] = (_Float16)x[((size_t)(b * NC + c)) * 16384 + p0 + p];
  }
  __syncthreads();
  for (int t = wave * 8; t < wave * 8 + 8; ++t) {       // 4m x 16n tiles
    int mt = t >> 4, nt = t & 15;
    v8f acc = {};
    wmma_rr(A16 + mt * 16 * 64, 64, Bt16 + nt * 16 * 64, 64, 2, acc);
    int n = lane & 15, rhi = (lane >> 4) << 3;
#pragma unroll
    for (int j = 0; j < 8; ++j) {
      int o = mt * 16 + j + rhi;
      X1[((size_t)(b * NC + o)) * 16384 + p0 + nt * 16 + n] = acc[j] + pw_b[o];
    }
  }
}

// row real-DFT: XR = x @ Cos, XI = -(x @ Sin); twiddles generated on the fly
__global__ void k_rowfft(const float* x, float* XR, float* XI) {
  const int img = blockIdx.x;           // b*64+c
  const int tid = threadIdx.x, wave = tid >> 5, lane = tid & 31;
  alignas(32) __shared__ _Float16 A16[128 * 128];
  for (int i = tid; i < 128 * 128; i += 256)
    A16[i] = (_Float16)x[(size_t)img * 16384 + i];
  __syncthreads();
  const int mt = wave;                  // 8 h-tiles
  for (int nt = 0; nt < 5; ++nt) {      // wf tiles 0..79
    v8f ar = {}, ai = {};
    wmma_rt(A16 + mt * 16 * 128, 128, FCos(), nt * 16, 4, ar);
    wmma_rt(A16 + mt * 16 * 128, 128, FSin(), nt * 16, 4, ai);
    int n = lane & 15, rhi = (lane >> 4) << 3;
#pragma unroll
    for (int j = 0; j < 8; ++j) {
      int h = mt * 16 + j + rhi, wf = nt * 16 + n;
      size_t o = ((size_t)img * NHF + h) * WFP + wf;
      XR[o] = ar[j];  XI[o] = -ai[j];
    }
  }
}

// column complex DFT: FR = C@XR + S@XI ; FI = C@XI - S@XR
// image operands staged transposed: [wf][h]
__global__ void k_colfft(const float* XR, const float* XI, float* FR, float* FI) {
  const int img = blockIdx.x;
  const int tid = threadIdx.x, wave = tid >> 5, lane = tid & 31;
  alignas(32) __shared__ _Float16 R16t[WFP * 128];
  alignas(32) __shared__ _Float16 I16t[WFP * 128];
  for (int i = tid; i < 128 * WFP; i += 256) {
    int h = i / WFP, wf = i % WFP;      // consecutive tid -> consecutive wf
    R16t[wf * 128 + h] = (_Float16)XR[(size_t)img * 128 * WFP + i];
    I16t[wf * 128 + h] = (_Float16)XI[(size_t)img * 128 * WFP + i];
  }
  __syncthreads();
  const int mt = wave;
  for (int nt = 0; nt < 5; ++nt) {
    v8f fr = {}, aq = {}, bq = {};
    wmma_tr(FCos(), mt * 16, R16t + nt * 16 * 128, 128, 4, fr);
    wmma_tr(FSin(), mt * 16, I16t + nt * 16 * 128, 128, 4, fr);
    wmma_tr(FCos(), mt * 16, I16t + nt * 16 * 128, 128, 4, aq);
    wmma_tr(FSin(), mt * 16, R16t + nt * 16 * 128, 128, 4, bq);
    int n = lane & 15, rhi = (lane >> 4) << 3;
#pragma unroll
    for (int j = 0; j < 8; ++j) {
      int hf = mt * 16 + j + rhi, wf = nt * 16 + n;
      size_t o = ((size_t)img * NHF + hf) * WFP + wf;
      FR[o] = fr[j];  FI[o] = aq[j] - bq[j];
    }
  }
}

// fused LayerNorm(channel) + 1/sqrt(n) + delta/decay (softplus dot with W_dt)
__global__ void k_build_u(const float* FR, const float* FI,
                          const float* ln_g, const float* ln_b,
                          const float* W_dt, const float* b_dt, const float* A_log,
                          float* U, float* DELTA, float* DEC) {
  const int blk = blockIdx.x;           // 0..NR-1
  const int b = blk / LSEQ, l = blk % LSEQ;
  const int hf = l / NWF, wf = l % NWF;
  const int d = threadIdx.x;            // 0..127
  const int c = d & 63;
  const float* src = (d < 64) ? FR : FI;
  float v = src[((size_t)(b * NC + c) * NHF + hf) * WFP + wf];
  __shared__ float red[128];
  red[d] = v; __syncthreads();
  for (int s = 64; s > 0; s >>= 1) { if (d < s) red[d] += red[d + s]; __syncthreads(); }
  float mu = red[0] * (1.f / 128.f); __syncthreads();
  red[d] = (v - mu) * (v - mu); __syncthreads();
  for (int s = 64; s > 0; s >>= 1) { if (d < s) red[d] += red[d + s]; __syncthreads(); }
  float var = red[0] * (1.f / 128.f); __syncthreads();
  float u = ((v - mu) * rsqrtf(var + 1e-5f) * ln_g[d] + ln_b[d]) * INV_SQRT_N;
  size_t r = (size_t)b * LSEQ + l;
  U[r * ND + d] = u;
  red[d] = u * W_dt[d]; __syncthreads();
  for (int s = 64; s > 0; s >>= 1) { if (d < s) red[d] += red[d + s]; __syncthreads(); }
  if (d == 0) {
    float s = red[0] + b_dt[0];
    float dt = (s > 20.f) ? s : log1pf(__expf(s));
    DELTA[r] = dt;
    DEC[r]   = __expf(-dt * __expf(A_log[0]));
  }
}

// Bt = U @ W_B ; Ct = U @ W_C   (M=NR, K=128, N=64)
__global__ void k_proj(const float* U, const _Float16* wB16t, const _Float16* wC16t,
                       float* BTo, float* CTo) {
  const int blk = blockIdx.x;           // 520 blocks of 64 rows
  const int tid = threadIdx.x, wave = tid >> 5, lane = tid & 31;
  const size_t r0 = (size_t)blk * 64;
  alignas(32) __shared__ _Float16 A16[64 * 128];
  alignas(32) __shared__ _Float16 Bb[64 * 128];        // [n][k]
  alignas(32) __shared__ _Float16 Cb[64 * 128];
  for (int i = tid; i < 64 * 128; i += 256) {
    A16[i] = (_Float16)U[r0 * ND + i];
    Bb[i] = wB16t[i];  Cb[i] = wC16t[i];
  }
  __syncthreads();
  for (int t = wave * 4; t < wave * 4 + 4; ++t) {      // 2x(4m x 4n) tiles
    int which = t >> 4, tl = t & 15, mt = tl >> 2, nt = tl & 3;
    const _Float16* Bp = which ? Cb : Bb;
    v8f acc = {};
    wmma_rr(A16 + mt * 16 * 128, 128, Bp + nt * 16 * 128, 128, 4, acc);
    float* dst = which ? CTo : BTo;
    int n = lane & 15, rhi = (lane >> 4) << 3;
#pragma unroll
    for (int j = 0; j < 8; ++j)
      dst[(r0 + mt * 16 + j + rhi) * NN + nt * 16 + n] = acc[j];
  }
}

// Chunked SSD scan (Mamba2 style), chunk T=16, all chunk math on WMMA.
__global__ void k_scan(const float* U, const float* DEC, const float* DELTA,
                       const float* BT, const float* CT, const float* h_prev,
                       const float* D_skip, float* Y, float* h_out) {
  const int b = blockIdx.x;
  const int tid = threadIdx.x, wave = tid >> 5, lane = tid & 31;
  alignas(16) __shared__ float    hT[ND * NN];         // state, TRANSPOSED [d][n]
  alignas(32) __shared__ _Float16 Cc16[16 * NN];       // chunk C (A operand)
  alignas(32) __shared__ _Float16 Bc16[16 * NN];       // chunk B rows = B^T as [s][n]
  alignas(32) __shared__ _Float16 UcT16[ND * 32];      // chunk U transposed [d][t], t-padded
  alignas(32) __shared__ _Float16 Wm16[16 * 32];       // masked decay*G (A), K-padded
  alignas(32) __shared__ _Float16 BwT16[NN * 32];      // decay-weighted B^T (A), K-padded
  __shared__ float Gf[16 * 16];
  __shared__ float sDec[16], sDelta[16], sP[16];

  for (int i = tid; i < ND * NN; i += 256) {
    int d = i >> 6, n = i & 63;
    hT[i] = h_prev[(size_t)b * NN * ND + n * ND + d];
  }
  for (int i = tid; i < ND * 16; i += 256) UcT16[(i >> 4) * 32 + 16 + (i & 15)] = (_Float16)0.f;
  for (int i = tid; i < 16 * 16; i += 256) Wm16[(i >> 4) * 32 + 16 + (i & 15)] = (_Float16)0.f;
  for (int i = tid; i < NN * 16; i += 256) BwT16[(i >> 4) * 32 + 16 + (i & 15)] = (_Float16)0.f;
  __syncthreads();

  for (int ci = 0; ci < LSEQ / 16; ++ci) {
    const size_t rb = (size_t)b * LSEQ + ci * 16;
    if (tid < 16) { sDec[tid] = DEC[rb + tid]; sDelta[tid] = DELTA[rb + tid]; }
    // prefetch next chunk's rows while this one is consumed (global_prefetch_b8)
    if (ci + 1 < LSEQ / 16 && tid >= 32 && tid < 48) {
      const size_t rn = rb + 16 + (tid - 32);
      __builtin_prefetch(&U[rn * ND], 0, 1);
      __builtin_prefetch(&BT[rn * NN], 0, 1);
      __builtin_prefetch(&CT[rn * NN], 0, 1);
    }
    for (int i = tid; i < 16 * NN; i += 256) {
      int t = i >> 6, n = i & 63;
      Cc16[i] = (_Float16)CT[(rb + t) * NN + n];
      Bc16[i] = (_Float16)BT[(rb + t) * NN + n];
    }
    for (int i = tid; i < 16 * ND; i += 256) {
      int t = i >> 7, d = i & 127;                      // consecutive tid -> consecutive d
      UcT16[d * 32 + t] = (_Float16)U[(rb + t) * ND + d];
    }
    __syncthreads();
    if (tid == 0) { float p = 1.f; for (int t = 0; t < 16; ++t) { p *= sDec[t]; sP[t] = p; } }
    __syncthreads();
    if (wave == 0) {                       // G = Cc @ Bc^T  (16x16, K=64)
      v8f g = {};
      wmma_rr(Cc16, NN, Bc16, NN, 2, g);
      int n = lane & 15, rhi = (lane >> 4) << 3;
#pragma unroll
      for (int j = 0; j < 8; ++j) Gf[(j + rhi) * 16 + n] = g[j];
    }
    __syncthreads();
    {                                      // causal decay mask
      int tt = tid >> 4, ss = tid & 15;
      float w = (ss <= tt) ? Gf[tt * 16 + ss] * sDelta[ss] * (sP[tt] / sP[ss]) : 0.f;
      Wm16[tt * 32 + ss] = (_Float16)w;
    }
    float p15 = sP[15];
    for (int i = tid; i < NN * 16; i += 256) {
      int n = i >> 4, s = i & 15;
      BwT16[n * 32 + s] =
          (_Float16)(BT[(rb + s) * NN + n] * sDelta[s] * (p15 / sP[s]));
    }
    __syncthreads();
    {                                      // y = Wm@U + p_t*(Cc@h) + Dskip*u
      const int dt0 = wave * 16;
      v8f yi = {}, ye = {};
      wmma_rr(Wm16, 32, UcT16 + dt0 * 32, 32, 1, yi);
      wmma_rf(Cc16, NN, hT + dt0 * NN, NN, 2, ye);
      int n = lane & 15, rhi = (lane >> 4) << 3;
      int d = dt0 + n;
      float dsk = D_skip[d];
#pragma unroll
      for (int j = 0; j < 8; ++j) {
        int t = j + rhi;
        size_t r = rb + t;
        Y[r * ND + d] = yi[j] + sP[t] * ye[j] + dsk * U[r * ND + d];
      }
    }
    __syncthreads();
    {                                      // hT <- p15*hT + (Bw^T @ U)^T
      int n0 = lane & 15, rhi = (lane >> 4) << 3;
      v8f hacc[4];
#pragma unroll
      for (int q = 0; q < 4; ++q) {
        int tl = wave * 4 + q, mt = tl >> 3, dt = tl & 7;
        v8f a = {};
        wmma_rr(BwT16 + mt * 16 * 32, 32, UcT16 + dt * 16 * 32, 32, 1, a);
        hacc[q] = a;
      }
#pragma unroll
      for (int q = 0; q < 4; ++q) {
        int tl = wave * 4 + q, mt = tl >> 3, dt = tl & 7;
#pragma unroll
        for (int j = 0; j < 8; ++j) {
          int row = mt * 16 + j + rhi;                  // state index n
          int col = dt * 16 + n0;                       // model index d
          hT[col * NN + row] = p15 * hT[col * NN + row] + hacc[q][j];
        }
      }
    }
    __syncthreads();
  }
  for (int i = tid; i < NN * ND; i += 256) {            // h_out is [b][n][d]
    int n = i >> 7, d = i & 127;
    h_out[(size_t)b * NN * ND + i] = hT[d * NN + n];
  }
}

// freq_out = (Y @ W_out + b_out)*sqrt_n, scattered to real/imag spectra
__global__ void k_wout(const float* Y, const _Float16* wo16t, const float* b_out,
                       float* FOr, float* FOi) {
  const int blk = blockIdx.x;               // 520 blocks of 64 rows
  const int tid = threadIdx.x, wave = tid >> 5, lane = tid & 31;
  const size_t r0 = (size_t)blk * 64;
  alignas(32) __shared__ _Float16 A16[64 * 128];
  alignas(32) __shared__ _Float16 Bt16[128 * 128];      // [n][k]
  for (int i = tid; i < 64 * 128; i += 256) A16[i] = (_Float16)Y[r0 * ND + i];
  for (int i = tid; i < 128 * 128; i += 256) Bt16[i] = wo16t[i];
  __syncthreads();
  for (int t = wave * 4; t < wave * 4 + 4; ++t) {       // 4m x 8n tiles
    int mt = t >> 3, nt = t & 7;
    v8f acc = {};
    wmma_rr(A16 + mt * 16 * 128, 128, Bt16 + nt * 16 * 128, 128, 4, acc);
    int n = lane & 15, rhi = (lane >> 4) << 3;
#pragma unroll
    for (int j = 0; j < 8; ++j) {
      int r = (int)r0 + mt * 16 + j + rhi;
      int b = r / LSEQ, l = r % LSEQ;
      int hf = l / NWF, wf = l % NWF;
      int d = nt * 16 + n, c = d & 63;
      float val = (acc[j] + b_out[d]) * SQRT_N;
      float* dst = (d < 64) ? FOr : FOi;
      dst[((size_t)(b * NC + c) * NHF + hf) * WFP + wf] = val;
    }
  }
}

// inverse column DFT: ZR = C@Fr - S@Fi ; ZI = S@Fr + C@Fi   (B staged [wf][h])
__global__ void k_invH(const float* FOr, const float* FOi, float* ZR, float* ZI) {
  const int img = blockIdx.x;
  const int tid = threadIdx.x, wave = tid >> 5, lane = tid & 31;
  alignas(32) __shared__ _Float16 R16t[WFP * 128];
  alignas(32) __shared__ _Float16 I16t[WFP * 128];
  for (int i = tid; i < 128 * WFP; i += 256) {
    int h = i / WFP, wf = i % WFP;
    R16t[wf * 128 + h] = (_Float16)FOr[(size_t)img * 128 * WFP + i];
    I16t[wf * 128 + h] = (_Float16)FOi[(size_t)img * 128 * WFP + i];
  }
  __syncthreads();
  const int mt = wave;
  for (int nt = 0; nt < 5; ++nt) {
    v8f a1 = {}, a2 = {}, a3 = {};
    wmma_tr(FCos(), mt * 16, R16t + nt * 16 * 128, 128, 4, a1);  // C@Fr
    wmma_tr(FSin(), mt * 16, I16t + nt * 16 * 128, 128, 4, a2);  // S@Fi
    wmma_tr(FSin(), mt * 16, R16t + nt * 16 * 128, 128, 4, a3);  // S@Fr
    wmma_tr(FCos(), mt * 16, I16t + nt * 16 * 128, 128, 4, a3);  // += C@Fi
    int n = lane & 15, rhi = (lane >> 4) << 3;
#pragma unroll
    for (int j = 0; j < 8; ++j) {
      int h = mt * 16 + j + rhi, wf = nt * 16 + n;
      size_t o = ((size_t)img * NHF + h) * WFP + wf;
      ZR[o] = a1[j] - a2[j];  ZI[o] = a3[j];
    }
  }
}

// inverse row rDFT (half-spectrum, Hermitian weights): x2 = ZR@IWc - ZI@IWs
__global__ void k_invW(const float* ZR, const float* ZI, float* X2) {
  const int img = blockIdx.x;
  const int tid = threadIdx.x, wave = tid >> 5, lane = tid & 31;
  alignas(32) __shared__ _Float16 R16[128 * 96];
  alignas(32) __shared__ _Float16 I16[128 * 96];
  for (int i = tid; i < 128 * 96; i += 256) {
    int h = i / 96, wf = i % 96;
    float vr = 0.f, vi = 0.f;
    if (wf < NWF) {
      size_t o = ((size_t)img * NHF + h) * WFP + wf;
      vr = ZR[o]; vi = ZI[o];
    }
    R16[i] = (_Float16)vr;  I16[i] = (_Float16)vi;
  }
  __syncthreads();
  const int mt = wave;
  for (int nt = 0; nt < 8; ++nt) {
    v8f ac = {}, as = {};
    wmma_rt(R16 + mt * 16 * 96, 96, FIwc(), nt * 16, 3, ac);
    wmma_rt(I16 + mt * 16 * 96, 96, FIws(), nt * 16, 3, as);
    int n = lane & 15, rhi = (lane >> 4) << 3;
#pragma unroll
    for (int j = 0; j < 8; ++j) {
      int h = mt * 16 + j + rhi, w = nt * 16 + n;
      X2[(size_t)img * 16384 + h * NW + w] = ac[j] - as[j];
    }
  }
}

__global__ void k_bnstats(const float* X2, float* BNM, float* BNV) {
  const int c = blockIdx.x, tid = threadIdx.x;
  float s = 0.f, s2 = 0.f;
  for (int e = tid; e < NB * 16384; e += 256) {
    int b = e >> 14, p = e & 16383;
    float v = X2[((size_t)(b * NC + c)) * 16384 + p];
    s += v; s2 += v * v;
  }
  __shared__ float r1[256], r2[256];
  r1[tid] = s; r2[tid] = s2; __syncthreads();
  for (int k = 128; k > 0; k >>= 1) {
    if (tid < k) { r1[tid] += r1[tid + k]; r2[tid] += r2[tid + k]; }
    __syncthreads();
  }
  if (tid == 0) {
    float m = r1[0] * (1.f / 65536.f);
    BNM[c] = m;  BNV[c] = r2[0] * (1.f / 65536.f) - m * m;
  }
}

__global__ void k_bn_gelu(const float* X2, const float* BNM, const float* BNV,
                          const float* bn_g, const float* bn_b, float* G0) {
  size_t i = (size_t)blockIdx.x * 256 + threadIdx.x;
  if (i >= (size_t)NB * NC * 16384) return;
  int c = (int)((i >> 14) & 63);
  float xn = (X2[i] - BNM[c]) * rsqrtf(BNV[c] + 1e-5f) * bn_g[c] + bn_b[c];
  G0[i] = gelu_exact(xn);
}

// g1 = gelu(w1 @ g0 + b1)   (128x64 x 64xP)
__global__ void k_mlp1(const float* G0, const _Float16* w1_16, const float* b1,
                       float* G1) {
  const int b = blockIdx.y, p0 = blockIdx.x * 256;
  const int tid = threadIdx.x, wave = tid >> 5, lane = tid & 31;
  alignas(32) __shared__ _Float16 A16[128 * 64];
  alignas(32) __shared__ _Float16 Bt16[256 * 64];       // [p][c]
  for (int i = tid; i < 128 * 64; i += 256) A16[i] = w1_16[i];
  for (int i = tid; i < 64 * 256; i += 256) {
    int c = i >> 8, p = i & 255;
    Bt16[p * 64 + c] = (_Float16)G0[((size_t)(b * NC + c)) * 16384 + p0 + p];
  }
  __syncthreads();
  for (int t = wave * 16; t < wave * 16 + 16; ++t) {    // 8m x 16n tiles
    int mt = t >> 4, nt = t & 15;
    v8f acc = {};
    wmma_rr(A16 + mt * 16 * 64, 64, Bt16 + nt * 16 * 64, 64, 2, acc);
    int n = lane & 15, rhi = (lane >> 4) << 3;
#pragma unroll
    for (int j = 0; j < 8; ++j) {
      int o = mt * 16 + j + rhi;
      float v = acc[j] + b1[o];
      G1[((size_t)(b * 128 + o)) * 16384 + p0 + nt * 16 + n] = gelu_exact(v);
    }
  }
}

// post = w2 @ g1 + b2 ; out = (1-a)*x1 + a*(x2 + post)
__global__ void k_mlp2_out(const float* G1, const _Float16* w2_16, const float* b2,
                           const float* X1, const float* X2, const float* alpha,
                           float* OUT) {
  const int b = blockIdx.y, p0 = blockIdx.x * 128;
  const int tid = threadIdx.x, wave = tid >> 5, lane = tid & 31;
  alignas(32) __shared__ _Float16 A16[64 * 128];
  alignas(32) __shared__ _Float16 Bt16[128 * 128];      // [p][c]
  for (int i = tid; i < 64 * 128; i += 256) A16[i] = w2_16[i];
  for (int i = tid; i < 128 * 128; i += 256) {
    int c = i >> 7, p = i & 127;
    Bt16[p * 128 + c] = (_Float16)G1[((size_t)(b * 128 + c)) * 16384 + p0 + p];
  }
  __syncthreads();
  const float a = alpha[0];
  for (int t = wave * 4; t < wave * 4 + 4; ++t) {       // 4m x 8n tiles
    int mt = t >> 3, nt = t & 7;
    v8f acc = {};
    wmma_rr(A16 + mt * 16 * 128, 128, Bt16 + nt * 16 * 128, 128, 4, acc);
    int n = lane & 15, rhi = (lane >> 4) << 3;
#pragma unroll
    for (int j = 0; j < 8; ++j) {
      int o = mt * 16 + j + rhi;
      size_t idx = ((size_t)(b * NC + o)) * 16384 + p0 + nt * 16 + n;
      float post = acc[j] + b2[o];
      OUT[idx] = (1.f - a) * X1[idx] + a * (X2[idx] + post);
    }
  }
}

// ---------------------------------------------------------------------------
extern "C" void kernel_launch(void* const* d_in, const int* in_sizes, int n_in,
                              void* d_out, int out_size, void* d_ws, size_t ws_size,
                              hipStream_t stream) {
  (void)in_sizes; (void)n_in; (void)out_size; (void)ws_size;
  const float* x     = (const float*)d_in[0];
  const float* alpha = (const float*)d_in[1];
  const float* hprev = (const float*)d_in[2];
  const float* pw_w  = (const float*)d_in[3];
  const float* pw_b  = (const float*)d_in[4];
  const float* ln_g  = (const float*)d_in[5];
  const float* ln_b  = (const float*)d_in[6];
  const float* W_dt  = (const float*)d_in[7];
  const float* b_dt  = (const float*)d_in[8];
  const float* W_B   = (const float*)d_in[9];
  const float* W_C   = (const float*)d_in[10];
  const float* A_log = (const float*)d_in[11];
  const float* Dskip = (const float*)d_in[12];
  const float* W_out = (const float*)d_in[13];
  const float* b_out = (const float*)d_in[14];
  const float* bn_g  = (const float*)d_in[15];
  const float* bn_b  = (const float*)d_in[16];
  const float* w1    = (const float*)d_in[17];
  const float* b1    = (const float*)d_in[18];
  const float* w2    = (const float*)d_in[19];
  const float* b2    = (const float*)d_in[20];

  float* out   = (float*)d_out;                       // [B,C,H,W]
  float* h_out = out + (size_t)NB * NC * NH * NW;     // [B,N,D]

  char* wp = (char*)d_ws;
  auto allocB = [&](size_t bytes) -> char* {
    char* p = wp; wp += (bytes + 255) & ~((size_t)255); return p;
  };
  _Float16* pw16  = (_Float16*)allocB(64 * 64 * 2);
  _Float16* wB16t = (_Float16*)allocB(128 * 64 * 2);
  _Float16* wC16t = (_Float16*)allocB(128 * 64 * 2);
  _Float16* wo16t = (_Float16*)allocB(128 * 128 * 2);
  _Float16* w1_16 = (_Float16*)allocB(128 * 64 * 2);
  _Float16* w2_16 = (_Float16*)allocB(64 * 128 * 2);
  const size_t IMGF = (size_t)NB * NC * NHF * WFP;    // 2,621,440
  float* XR  = (float*)allocB(IMGF * 4);              // also reused as FOr
  float* XI  = (float*)allocB(IMGF * 4);              // also reused as FOi
  float* FR  = (float*)allocB(IMGF * 4);              // also reused as ZR
  float* FI  = (float*)allocB(IMGF * 4);              // also reused as ZI
  float* U   = (float*)allocB((size_t)NR * ND * 4);
  float* DLT = (float*)allocB((size_t)NR * 4);
  float* DEC = (float*)allocB((size_t)NR * 4);
  float* BT  = (float*)allocB((size_t)NR * NN * 4);
  float* CT  = (float*)allocB((size_t)NR * NN * 4);
  float* Yb  = (float*)allocB((size_t)NR * ND * 4);
  float* X1  = (float*)allocB((size_t)NB * NC * 16384 * 4);
  float* X2  = (float*)allocB((size_t)NB * NC * 16384 * 4);
  float* G0  = (float*)allocB((size_t)NB * NC * 16384 * 4);
  float* G1  = (float*)allocB((size_t)NB * 128 * 16384 * 4);
  float* BNM = (float*)allocB(64 * 4);
  float* BNV = (float*)allocB(64 * 4);

  k_prep<<<64, 256, 0, stream>>>(pw_w, W_B, W_C, W_out, w1, w2,
                                 pw16, wB16t, wC16t, wo16t, w1_16, w2_16);
  k_pw<<<dim3(64, NB), 256, 0, stream>>>(x, pw16, pw_b, X1);
  k_rowfft<<<NB * NC, 256, 0, stream>>>(x, XR, XI);
  k_colfft<<<NB * NC, 256, 0, stream>>>(XR, XI, FR, FI);
  k_build_u<<<NR, 128, 0, stream>>>(FR, FI, ln_g, ln_b, W_dt, b_dt, A_log,
                                    U, DLT, DEC);
  k_proj<<<NR / 64, 256, 0, stream>>>(U, wB16t, wC16t, BT, CT);
  k_scan<<<NB, 256, 0, stream>>>(U, DEC, DLT, BT, CT, hprev, Dskip, Yb, h_out);
  k_wout<<<NR / 64, 256, 0, stream>>>(Yb, wo16t, b_out, XR, XI);  // -> spectra
  k_invH<<<NB * NC, 256, 0, stream>>>(XR, XI, FR, FI);            // -> ZR/ZI
  k_invW<<<NB * NC, 256, 0, stream>>>(FR, FI, X2);
  k_bnstats<<<NC, 256, 0, stream>>>(X2, BNM, BNV);
  k_bn_gelu<<<16384, 256, 0, stream>>>(X2, BNM, BNV, bn_g, bn_b, G0);
  k_mlp1<<<dim3(64, NB), 256, 0, stream>>>(G0, w1_16, b1, G1);
  k_mlp2_out<<<dim3(128, NB), 256, 0, stream>>>(G1, w2_16, b2, X1, X2, alpha, out);
}
